// Simple_spliter_62569083568141
// MI455X (gfx1250) — matrix operand order repaired and checked by reading, not compiled, not run on hardware
//
#include <hip/hip_runtime.h>
#include <hip/hip_bf16.h>

typedef __attribute__((ext_vector_type(16))) _Float16 v16h;
typedef __attribute__((ext_vector_type(8)))  float    v8f;

#define EDIM   1280
#define BATCH  1024
#define SEQ    77
#define K1PAD  96
#define NW     4     // waves per block
#define TPW    8     // tiles per wave
#define ETILES 80    // EDIM / 16

union AFrag {
  v16h h16;
  _Float16 h[16];
  uint4 q[2];
};

__device__ inline float eluf(float x) { return x > 0.0f ? x : expf(x) - 1.0f; }

__device__ inline v8f wmma16(v16h a, v16h b, v8f c) {
  return __builtin_amdgcn_wmma_f32_16x16x32_f16(false, a, false, b, (short)0, c,
                                                false, false);
}

// B fragment stored pre-swizzled in LDS as [c(2)][lane(32)][8] f16 (512 f16 per frag)
__device__ inline v16h load_bfrag(const _Float16* frag, int lane) {
  AFrag f;
  f.q[0] = *reinterpret_cast<const uint4*>(frag + lane * 8);
  f.q[1] = *reinterpret_cast<const uint4*>(frag + 256 + lane * 8);
  return f.h16;
}

// A fragment from row-major f16 LDS tile [16][ldk]; ISA 16-bit A 16x32 layout:
// lane<16: K = k0+{0..7, 16..23}; lane>=16: K = k0+{8..15, 24..31}, row = lane&15
__device__ inline v16h load_afrag(const _Float16* Ab, int ldk, int lane, int k0) {
  const _Float16* p = Ab + (lane & 15) * ldk + k0 + ((lane >> 4) << 3);
  AFrag f;
  f.q[0] = *reinterpret_cast<const uint4*>(p);
  f.q[1] = *reinterpret_cast<const uint4*>(p + 16);
  return f.h16;
}

__device__ inline v16h scale_frag(v16h a, _Float16 s) {
  AFrag f; f.h16 = a;
#pragma unroll
  for (int i = 0; i < 16; ++i) f.h[i] *= s;
  return f.h16;
}

// Convert global f32 weight [din][dout] into LDS B-fragment layout.
__device__ inline void fill_frag(_Float16* dst, const float* Wg, int din, int dout,
                                 int ksteps, int ntiles, int tid, int nth) {
  int total = ksteps * ntiles * 512;
  for (int i = tid; i < total; i += nth) {
    int t = i / (ntiles * 512);
    int r = i - t * (ntiles * 512);
    int j = r >> 9;
    int q = r & 511;
    int c = q >> 8;
    int lane = (q >> 3) & 31;
    int w = q & 7;
    int k = t * 32 + ((lane >> 4) << 3) + c * 16 + w;
    int n = (j << 4) + (lane & 15);
    float val = (k < din) ? Wg[(size_t)k * dout + n] : 0.0f;
    dst[i] = (_Float16)val;
  }
}

// D (f32 16x16) layout: lane holds N=lane&15, rows M = (lane>>4)*8 + r
__device__ inline void store_tile_f16(v8f c, _Float16* dst, int ldn, int n0,
                                      const float* bias, int lane) {
  int n = (lane & 15) + n0;
  int mb = (lane >> 4) * 8;
  float bv = bias[n];
#pragma unroll
  for (int r = 0; r < 8; ++r)
    dst[(mb + r) * ldn + n] = (_Float16)eluf(c[r] + bv);
}

__device__ inline void store_tile_f32(v8f c, float* dst, int ldn, int n0,
                                      const float* bias, int lane, bool do_elu) {
  int n = (lane & 15) + n0;
  int mb = (lane >> 4) * 8;
  float bv = bias[n];
#pragma unroll
  for (int r = 0; r < 8; ++r) {
    float v = c[r] + bv;
    dst[(mb + r) * ldn + n] = do_elu ? eluf(v) : v;
  }
}

struct SharedMem {
  _Float16 w1[3 * 8 * 512];   // 24 KB, K padded 79->96 with zeros
  _Float16 w2[4 * 4 * 512];   // 16 KB
  _Float16 w3[2 * 2 * 512];   // 4 KB
  _Float16 w4[1 * 1 * 512];   // 1 KB
  float w5[16 * 8];
  float w6[8 * 4];
  float w7[4 * 2];
  float w8[2 * 1];
  float b1[128]; float b2[64]; float b3[32]; float b4[16];
  float b5[8];   float b6[4];  float b7[2];  float b8[1];
  alignas(16) char buf[NW][4096];   // per-wave staging buffer
};

// ---------------- Kernel A: pre_out GEMM + increment -----------------------
__global__ void __launch_bounds__(128) prehead_kernel(
    const float* __restrict__ prior, const float* __restrict__ W0,
    const float* __restrict__ b0, const float* __restrict__ rise,
    const float* __restrict__ Winc, const float* __restrict__ binc,
    float* __restrict__ Pre, float* __restrict__ Inc) {
  int wave = blockIdx.x * (blockDim.x >> 5) + (threadIdx.x >> 5);
  int lane = threadIdx.x & 31;
  int bt = wave / ETILES;
  int et = wave - bt * ETILES;
  int b0r = bt << 4, e0 = et << 4;
  int m = lane & 15, hf = lane >> 4;
  v8f c = {};
  for (int k0 = 0; k0 < EDIM; k0 += 32) {
    AFrag a, bf;
    const float* pa = prior + (size_t)(b0r + m) * EDIM + k0 + hf * 16;
#pragma unroll
    for (int i = 0; i < 8; ++i) {
      a.h[i]     = (_Float16)pa[i];        // K = k0 + hf*16 + i
      a.h[8 + i] = (_Float16)pa[8 + i];    // K = k0 + hf*16 + 8 + i
    }
    const float* pw = W0 + (size_t)(k0 + hf * 16) * EDIM + e0 + m;
#pragma unroll
    for (int i = 0; i < 16; ++i)
      bf.h[i] = (_Float16)pw[(size_t)i * EDIM];   // K consecutive, N = e0+m
    c = wmma16(a.h16, bf.h16, c);
  }
  int n = e0 + m;
  float bias = b0[n], wi = Winc[n], bi = binc[n];
  int mb = hf * 8;
#pragma unroll
  for (int r = 0; r < 8; ++r) {
    int brow = b0r + mb + r;
    Pre[(size_t)brow * EDIM + n] = c[r] + bias;
    Inc[(size_t)brow * EDIM + n] = rise[brow] * wi + bi;
  }
}

// ---------------- Kernel B: normalize + funnel -----------------------------
__global__ void __launch_bounds__(NW * 32) funnel_kernel(
    const float* __restrict__ text, const float* __restrict__ Pre,
    const float* __restrict__ Inc,
    const float* __restrict__ W1g, const float* __restrict__ B1g,
    const float* __restrict__ W2g, const float* __restrict__ B2g,
    const float* __restrict__ W3g, const float* __restrict__ B3g,
    const float* __restrict__ W4g, const float* __restrict__ B4g,
    const float* __restrict__ W5g, const float* __restrict__ B5g,
    const float* __restrict__ W6g, const float* __restrict__ B6g,
    const float* __restrict__ W7g, const float* __restrict__ B7g,
    const float* __restrict__ W8g, const float* __restrict__ B8g,
    float* __restrict__ out) {
  __shared__ SharedMem sh;
  const int tid = threadIdx.x, nth = blockDim.x;

  fill_frag(sh.w1, W1g, 79, 128, 3, 8, tid, nth);
  fill_frag(sh.w2, W2g, 128, 64, 4, 4, tid, nth);
  fill_frag(sh.w3, W3g, 64, 32, 2, 2, tid, nth);
  fill_frag(sh.w4, W4g, 32, 16, 1, 1, tid, nth);
  for (int i = tid; i < 128; i += nth) sh.w5[i] = W5g[i];
  for (int i = tid; i < 32;  i += nth) sh.w6[i] = W6g[i];
  for (int i = tid; i < 8;   i += nth) sh.w7[i] = W7g[i];
  for (int i = tid; i < 2;   i += nth) sh.w8[i] = W8g[i];
  for (int i = tid; i < 128; i += nth) sh.b1[i] = B1g[i];
  for (int i = tid; i < 64;  i += nth) sh.b2[i] = B2g[i];
  for (int i = tid; i < 32;  i += nth) sh.b3[i] = B3g[i];
  for (int i = tid; i < 16;  i += nth) sh.b4[i] = B4g[i];
  for (int i = tid; i < 8;   i += nth) sh.b5[i] = B5g[i];
  for (int i = tid; i < 4;   i += nth) sh.b6[i] = B6g[i];
  for (int i = tid; i < 2;   i += nth) sh.b7[i] = B7g[i];
  for (int i = tid; i < 1;   i += nth) sh.b8[i] = B8g[i];
  __syncthreads();

  const int lane = tid & 31;
  const int wv = tid >> 5;
  char* base = sh.buf[wv];
  _Float16* ab16 = reinterpret_cast<_Float16*>(base);
  const int m = lane & 15;
  const int hf = lane >> 4;

  for (int t = 0; t < TPW; ++t) {
    int g = blockIdx.x * (NW * TPW) + wv * TPW + t;
    int bb = g / ETILES;
    int e0 = (g - bb * ETILES) << 4;
    int e = e0 + m;

    // ---- stage the 79-row (f16) and compute L2 norm over axis=1 ----
    float ssq = 0.0f;
    const float* tp = text + (size_t)bb * SEQ * EDIM + e;
    for (int s = hf; s < SEQ; s += 2) {
      float v = tp[(size_t)s * EDIM];
      ab16[m * K1PAD + s] = (_Float16)v;
      ssq += v * v;
    }
    {
      float v = hf ? Inc[(size_t)bb * EDIM + e] : Pre[(size_t)bb * EDIM + e];
      ab16[m * K1PAD + 77 + hf] = (_Float16)v;   // row77=pre_out, row78=increment
      ssq += v * v;
    }
    for (int k = 79 + hf; k < K1PAD; k += 2) ab16[m * K1PAD + k] = (_Float16)0.0f;
    float tot = ssq + __shfl_xor(ssq, 16, 32);
    float scale = 1.0f / fmaxf(sqrtf(tot), 1e-12f);
    __syncthreads();

    // ---- layer 1: [16,96] x [96,128], ELU -> f16 [16][128] @0 ----
    _Float16 sc = (_Float16)scale;
    v16h a0 = scale_frag(load_afrag(ab16, K1PAD, lane, 0), sc);
    v16h a1 = scale_frag(load_afrag(ab16, K1PAD, lane, 32), sc);
    v16h a2 = scale_frag(load_afrag(ab16, K1PAD, lane, 64), sc);
#pragma unroll
    for (int j = 0; j < 8; ++j) {
      v8f c = {};
      c = wmma16(a0, load_bfrag(sh.w1 + (0 * 8 + j) * 512, lane), c);
      c = wmma16(a1, load_bfrag(sh.w1 + (1 * 8 + j) * 512, lane), c);
      c = wmma16(a2, load_bfrag(sh.w1 + (2 * 8 + j) * 512, lane), c);
      store_tile_f16(c, ab16, 128, j * 16, sh.b1, lane);
    }
    __syncthreads();

    // ---- layer 2: [16,128] x [128,64], ELU -> f16 [16][64] @0 ----
    v16h x0 = load_afrag(ab16, 128, lane, 0);
    v16h x1 = load_afrag(ab16, 128, lane, 32);
    v16h x2 = load_afrag(ab16, 128, lane, 64);
    v16h x3 = load_afrag(ab16, 128, lane, 96);
#pragma unroll
    for (int j = 0; j < 4; ++j) {
      v8f c = {};
      c = wmma16(x0, load_bfrag(sh.w2 + (0 * 4 + j) * 512, lane), c);
      c = wmma16(x1, load_bfrag(sh.w2 + (1 * 4 + j) * 512, lane), c);
      c = wmma16(x2, load_bfrag(sh.w2 + (2 * 4 + j) * 512, lane), c);
      c = wmma16(x3, load_bfrag(sh.w2 + (3 * 4 + j) * 512, lane), c);
      store_tile_f16(c, ab16, 64, j * 16, sh.b2, lane);
    }
    __syncthreads();

    // ---- layer 3: [16,64] x [64,32] + bias (no act) -> f32 [16][32] @+2048 ----
    float* f3 = reinterpret_cast<float*>(base + 2048);
    {
      v16h y0 = load_afrag(ab16, 64, lane, 0);
      v16h y1 = load_afrag(ab16, 64, lane, 32);
#pragma unroll
      for (int j = 0; j < 2; ++j) {
        v8f c = {};
        c = wmma16(y0, load_bfrag(sh.w3 + (0 * 2 + j) * 512, lane), c);
        c = wmma16(y1, load_bfrag(sh.w3 + (1 * 2 + j) * 512, lane), c);
        store_tile_f32(c, f3, 32, j * 16, sh.b3, lane, false);
      }
    }
    __syncthreads();

    // ---- LayerNorm(32) + ELU -> f16 [16][32] @0 ----
    {
      float vr[16];
      const float4* p4 = reinterpret_cast<const float4*>(f3 + m * 32 + hf * 16);
#pragma unroll
      for (int i = 0; i < 4; ++i) {
        float4 q = p4[i];
        vr[4 * i + 0] = q.x; vr[4 * i + 1] = q.y;
        vr[4 * i + 2] = q.z; vr[4 * i + 3] = q.w;
      }
      float ps = 0;
#pragma unroll
      for (int i = 0; i < 16; ++i) ps += vr[i];
      float mean = (ps + __shfl_xor(ps, 16, 32)) * (1.0f / 32.0f);
      float pq = 0;
#pragma unroll
      for (int i = 0; i < 16; ++i) { float d = vr[i] - mean; pq += d * d; }
      float inv = rsqrtf((pq + __shfl_xor(pq, 16, 32)) * (1.0f / 32.0f) + 1e-5f);
#pragma unroll
      for (int i = 0; i < 16; ++i)
        ab16[m * 32 + hf * 16 + i] = (_Float16)eluf((vr[i] - mean) * inv);
    }
    __syncthreads();

    // ---- layer 4: [16,32] x [32,16] + bias + ELU -> f32 [16][16] @+2048 ----
    float* f4 = reinterpret_cast<float*>(base + 2048);
    {
      v16h z0 = load_afrag(ab16, 32, lane, 0);
      v8f c = {};
      c = wmma16(z0, load_bfrag(sh.w4, lane), c);
      store_tile_f32(c, f4, 16, 0, sh.b4, lane, true);
    }
    __syncthreads();

    // ---- layers 5..8 per row (lanes 0..15, one row each) ----
    if (lane < 16) {
      float v[16];
#pragma unroll
      for (int i = 0; i < 16; ++i) v[i] = f4[lane * 16 + i];
      float h5[8];
#pragma unroll
      for (int o = 0; o < 8; ++o) {
        float s = sh.b5[o];
#pragma unroll
        for (int i = 0; i < 16; ++i) s += v[i] * sh.w5[i * 8 + o];
        h5[o] = s;
      }
      float mn = 0;
#pragma unroll
      for (int o = 0; o < 8; ++o) mn += h5[o];
      mn *= 0.125f;
      float vq = 0;
#pragma unroll
      for (int o = 0; o < 8; ++o) { float d = h5[o] - mn; vq += d * d; }
      float inv = rsqrtf(vq * 0.125f + 1e-5f);
#pragma unroll
      for (int o = 0; o < 8; ++o) h5[o] = eluf((h5[o] - mn) * inv);
      float h6[4];
#pragma unroll
      for (int o = 0; o < 4; ++o) {
        float s = sh.b6[o];
#pragma unroll
        for (int i = 0; i < 8; ++i) s += h5[i] * sh.w6[i * 4 + o];
        h6[o] = eluf(s);
      }
      float h7[2];
#pragma unroll
      for (int o = 0; o < 2; ++o) {
        float s = sh.b7[o];
#pragma unroll
        for (int i = 0; i < 4; ++i) s += h6[i] * sh.w7[i * 2 + o];
        h7[o] = eluf(s);
      }
      float r8 = sh.b8[0] + h7[0] * sh.w8[0] + h7[1] * sh.w8[1];
      out[(size_t)bb * EDIM + e0 + lane] = r8;
    }
    __syncthreads();
  }
}

extern "C" void kernel_launch(void* const* d_in, const int* in_sizes, int n_in,
                              void* d_out, int out_size, void* d_ws, size_t ws_size,
                              hipStream_t stream) {
  const float* text  = (const float*)d_in[0];
  const float* prior = (const float*)d_in[1];
  const float* rise  = (const float*)d_in[2];
  const float* Winc  = (const float*)d_in[3];
  const float* binc  = (const float*)d_in[4];
  const float* W0    = (const float*)d_in[5];
  const float* b0    = (const float*)d_in[6];
  const float* W[8];
  const float* Bv[8];
  for (int i = 0; i < 8; ++i) {
    W[i]  = (const float*)d_in[7 + 2 * i];
    Bv[i] = (const float*)d_in[8 + 2 * i];
  }
  float* out = (float*)d_out;
  float* Pre = (float*)d_ws;                       // [B*E] f32
  float* Inc = Pre + (size_t)BATCH * EDIM;         // [B*E] f32

  // 64 b-tiles * 80 e-tiles = 5120 waves, 4 waves/block
  prehead_kernel<<<(BATCH / 16) * ETILES / 4, 128, 0, stream>>>(
      prior, W0, b0, rise, Winc, binc, Pre, Inc);

  // 81920 tiles / (4 waves * 8 tiles) = 2560 blocks
  funnel_kernel<<<(BATCH * ETILES) / (NW * TPW), NW * 32, 0, stream>>>(
      text, Pre, Inc,
      W[0], Bv[0], W[1], Bv[1], W[2], Bv[2], W[3], Bv[3],
      W[4], Bv[4], W[5], Bv[5], W[6], Bv[6], W[7], Bv[7], out);
}
